// Llama_51874615001305
// MI455X (gfx1250) — matrix operand order repaired and checked by reading, not compiled, not run on hardware
//
#include <hip/hip_runtime.h>
#include <hip/hip_bf16.h>
#include <stdint.h>

// ---------------- model dims ----------------
constexpr int kL   = 8;
constexpr int kDim = 1024;
constexpr int kH   = 16;
constexpr int kHd  = 64;
constexpr int kFF  = 4096;
constexpr int kLat = 256;
constexpr int kB   = 4;
constexpr int kS   = 512;
constexpr int kSc  = 64;
constexpr int kT   = kSc + 1 + kS;   // 577
constexpr int kBT  = kB * kT;        // 2308

typedef __attribute__((ext_vector_type(16))) __bf16 v16bf;
typedef __attribute__((ext_vector_type(8)))  float  v8f;

__device__ __forceinline__ unsigned short f2bf(float f) {
  unsigned int u = __float_as_uint(f);
  u += 0x7FFFu + ((u >> 16) & 1u);   // round-to-nearest-even
  return (unsigned short)(u >> 16);
}

__device__ __forceinline__ void wait_asynccnt0() {
#if __has_builtin(__builtin_amdgcn_s_wait_asynccnt)
  __builtin_amdgcn_s_wait_asynccnt(0);
#else
  asm volatile("s_wait_asynccnt 0x0" ::: "memory");
#endif
}

// ---------------------------------------------------------------------------
// Weight prep: W[K,N] fp32 -> Wt[N,K] bf16 (transpose + convert), 32x32 tiles
// ---------------------------------------------------------------------------
__global__ __launch_bounds__(256) void transpose_bf16_kernel(
    const float* __restrict__ W, unsigned short* __restrict__ Wt, int K, int N)
{
  __shared__ float tile[32][33];
  const int n0 = blockIdx.x * 32, k0 = blockIdx.y * 32;
  const int tx = threadIdx.x & 31, ty = threadIdx.x >> 5;   // 32 x 8
#pragma unroll
  for (int i = 0; i < 32; i += 8)
    tile[ty + i][tx] = W[(long long)(k0 + ty + i) * N + n0 + tx];
  __syncthreads();
#pragma unroll
  for (int i = 0; i < 32; i += 8)
    Wt[(long long)(n0 + ty + i) * K + k0 + tx] = f2bf(tile[tx][ty + i]);
}

// ---------------------------------------------------------------------------
// GEMM: C[M,N] = epilogue(A[M,K] @ B[K,N]); A fp32, B given as Bt[N,K] bf16.
// flags: 1 = add bias, 2 = SiLU, 4 = add residual Res
// Block tile 128x128, K-step 32, 256 threads = 8 waves (4M x 2N),
// each wave 32x64 = 8 x v_wmma_f32_16x16x32_bf16 per K-step.
// B tile staged via global_load_async_to_lds_b128 (ASYNCcnt).
// grid.z = batch; A/C (and Res) advance by sA/sC per batch.
// ---------------------------------------------------------------------------
__global__ __launch_bounds__(256) void gemm_bf16_kernel(
    const float* __restrict__ A, const unsigned short* __restrict__ Bt,
    const float* __restrict__ bias, const float* __restrict__ Res,
    float* __restrict__ C, int M, int N, int K,
    long long sA, long long sC, int flags)
{
  A += (long long)blockIdx.z * sA;
  C += (long long)blockIdx.z * sC;
  if (Res) Res += (long long)blockIdx.z * sC;

  const int Mbase = blockIdx.y * 128;
  const int Nbase = blockIdx.x * 128;

  // bf16 LDS tiles, padded stride 40 shorts (80B rows -> 16B-aligned b128 ops)
  __shared__ __align__(16) unsigned short As[128 * 40];
  __shared__ __align__(16) unsigned short Bs[128 * 40];

  const int tid  = threadIdx.x;
  const int lane = tid & 31;
  const int w    = tid >> 5;
  const int wm   = (w & 3) * 32;   // wave M offset within block
  const int wn   = (w >> 2) * 64;  // wave N offset within block
  const int hl   = lane >> 4;      // lane half (0/1)
  const int m16  = lane & 15;

  v8f acc[2][4];
#pragma unroll
  for (int a = 0; a < 2; ++a)
#pragma unroll
    for (int b = 0; b < 4; ++b)
#pragma unroll
      for (int r = 0; r < 8; ++r) acc[a][b][r] = 0.f;

  for (int kb = 0; kb < K; kb += 32) {
    __syncthreads();   // previous iteration's fragment reads complete

    // ---- B tile: 32(K) x 128(N) bf16, async DMA global->LDS, transposed
    //      layout in LDS: Bs[n][k], 16B per lane per op, 512 ops total ----
#pragma unroll
    for (int it = 0; it < 2; ++it) {
      int lin = tid + it * 256;          // 0..511
      int n   = lin & 127;
      int kc  = lin >> 7;                // 0..3 (8 bf16 each)
      const unsigned short* gp = Bt + (long long)(Nbase + n) * K + kb + kc * 8;
      unsigned lds = (unsigned)(uintptr_t)&Bs[n * 40 + kc * 8];
      asm volatile("global_load_async_to_lds_b128 %0, %1, off"
                   :: "v"(lds), "v"(gp) : "memory");
    }

    // ---- A tile: 128x32 fp32 -> bf16 LDS (16 elems / thread) ----
#pragma unroll
    for (int it = 0; it < 4; ++it) {
      int idx = tid * 16 + it * 4;
      int r = idx >> 5, c = idx & 31;
      int gr = Mbase + r;
      float4 f;
      if (gr < M) f = *reinterpret_cast<const float4*>(A + (long long)gr * K + kb + c);
      else        f = make_float4(0.f, 0.f, 0.f, 0.f);
      union { unsigned short s[4]; uint2 u; } pk;
      pk.s[0] = f2bf(f.x); pk.s[1] = f2bf(f.y);
      pk.s[2] = f2bf(f.z); pk.s[3] = f2bf(f.w);
      *reinterpret_cast<uint2*>(&As[r * 40 + c]) = pk.u;
    }

    wait_asynccnt0();   // this wave's async DMA landed in LDS
    __syncthreads();    // all waves' tiles visible

    // ---- fragments per documented wave32 layouts ----
    union Frag { uint4 u[2]; v16bf v; };
    Frag fa[2], fb[4];
#pragma unroll
    for (int a = 0; a < 2; ++a) {
      int r = wm + a * 16 + m16;                    // A row (M)
      const unsigned short* p = &As[r * 40 + hl * 8];
      fa[a].u[0] = *reinterpret_cast<const uint4*>(p);       // K = hl*8 + 0..7
      fa[a].u[1] = *reinterpret_cast<const uint4*>(p + 16);  // K = 16 + hl*8 + 0..7
    }
#pragma unroll
    for (int b = 0; b < 4; ++b) {
      int cc = wn + b * 16 + m16;                   // B col (N)
      const unsigned short* p = &Bs[cc * 40 + hl * 8];
      fb[b].u[0] = *reinterpret_cast<const uint4*>(p);       // K = hl*8 + 0..7
      fb[b].u[1] = *reinterpret_cast<const uint4*>(p + 16);  // K = 16 + hl*8 + 0..7
    }
#pragma unroll
    for (int a = 0; a < 2; ++a)
#pragma unroll
      for (int b = 0; b < 4; ++b)
        acc[a][b] = __builtin_amdgcn_wmma_f32_16x16x32_bf16(
            false, fa[a].v, false, fb[b].v, (short)0, acc[a][b], false, false);
  }

  // ---- epilogue ----
#pragma unroll
  for (int a = 0; a < 2; ++a)
#pragma unroll
    for (int b = 0; b < 4; ++b) {
      int col = Nbase + wn + b * 16 + m16;
      float bv = (flags & 1) ? bias[col] : 0.f;
#pragma unroll
      for (int r = 0; r < 8; ++r) {
        int row = Mbase + wm + a * 16 + hl * 8 + r;
        if (row < M) {
          float v = acc[a][b][r] + bv;
          if (flags & 2) v = v / (1.f + __expf(-v));            // SiLU
          if (flags & 4) v += Res[(long long)row * N + col];    // residual
          C[(long long)row * N + col] = v;
        }
      }
    }
}

// ---------------------------------------------------------------------------
// RMSNorm: one row (DIM=1024) per block of 256 threads
// ---------------------------------------------------------------------------
__global__ __launch_bounds__(256) void rmsnorm_kernel(
    const float* __restrict__ x, const float* __restrict__ w, float* __restrict__ y)
{
  const long long row = blockIdx.x;
  const float* xr = x + row * kDim;
  float4 f = *reinterpret_cast<const float4*>(xr + threadIdx.x * 4);
  float ss = f.x * f.x + f.y * f.y + f.z * f.z + f.w * f.w;
#pragma unroll
  for (int o = 16; o > 0; o >>= 1) ss += __shfl_xor(ss, o, 32);
  __shared__ float red[8];
  if ((threadIdx.x & 31) == 0) red[threadIdx.x >> 5] = ss;
  __syncthreads();
  float tot = 0.f;
#pragma unroll
  for (int i = 0; i < 8; ++i) tot += red[i];
  float inv = rsqrtf(tot * (1.f / (float)kDim) + 1e-6f);
  float4 wv = *reinterpret_cast<const float4*>(w + threadIdx.x * 4);
  float4 o;
  o.x = f.x * inv * wv.x; o.y = f.y * inv * wv.y;
  o.z = f.z * inv * wv.z; o.w = f.w * inv * wv.w;
  *reinterpret_cast<float4*>(y + row * kDim + threadIdx.x * 4) = o;
}

// ---------------------------------------------------------------------------
// Timestep embedding: te0[b,d] = cos/sin(t[b] * freq)
// ---------------------------------------------------------------------------
__global__ void timeembed_kernel(const float* __restrict__ t, float* __restrict__ te0)
{
  int b = blockIdx.x;
  float tv = t[b];
  for (int d = threadIdx.x; d < kDim; d += blockDim.x) {
    int hd = (d < kDim / 2) ? d : d - kDim / 2;
    float fr = __expf(-9.210340371976184f * (float)hd / (float)(kDim / 2)); // ln(1e4)
    float a = tv * fr;
    te0[b * kDim + d] = (d < kDim / 2) ? cosf(a) : sinf(a);
  }
}

// ---------------------------------------------------------------------------
// SwiGLU gate: g = silu(g) * c, in place on g  (n multiple of 4)
// ---------------------------------------------------------------------------
__global__ void gate_kernel(float* __restrict__ g, const float* __restrict__ c, int n)
{
  int i = (blockIdx.x * blockDim.x + threadIdx.x) * 4;
  if (i < n) {
    float4 a = *reinterpret_cast<float4*>(g + i);
    float4 b = *reinterpret_cast<const float4*>(c + i);
    a.x = a.x / (1.f + __expf(-a.x)) * b.x;
    a.y = a.y / (1.f + __expf(-a.y)) * b.y;
    a.z = a.z / (1.f + __expf(-a.z)) * b.z;
    a.w = a.w / (1.f + __expf(-a.w)) * b.w;
    *reinterpret_cast<float4*>(g + i) = a;
  }
}

// ---------------------------------------------------------------------------
// Causal attention: grid (B, H, ceil(T/8)); 8 waves/block, 1 query per wave.
// Q/K/V layout: [(b*T + pos)*DIM + head*64 + d], fp32.
// ---------------------------------------------------------------------------
__global__ __launch_bounds__(256) void attn_kernel(
    const float* __restrict__ Q, const float* __restrict__ Kv,
    const float* __restrict__ V, float* __restrict__ O, int T)
{
  const int b  = blockIdx.x;
  const int hh = blockIdx.y;
  const int w    = threadIdx.x >> 5;
  const int lane = threadIdx.x & 31;
  const int qi   = blockIdx.z * 8 + w;
  const bool active = (qi < T);
  const int qc = active ? qi : (T - 1);

  __shared__ float sp[8][608];   // score row per wave (T=577 <= 608)

  float qreg[64];
  const long long qbase = ((long long)(b * T + qc)) * kDim + hh * kHd;
#pragma unroll
  for (int d = 0; d < 64; d += 2) {
    float2 f = *reinterpret_cast<const float2*>(Q + qbase + d);
    qreg[d] = f.x; qreg[d + 1] = f.y;
  }

  const int KB = min(T, blockIdx.z * 8 + 8);  // uniform bound across block
  const float scale = 0.125f;                 // 1/sqrt(64)

  // pass 1: scores
  for (int k0 = lane; k0 < KB; k0 += 32) {
    const float* kr = Kv + ((long long)(b * T + k0)) * kDim + hh * kHd;
    float dot = 0.f;
#pragma unroll
    for (int d = 0; d < 64; ++d) dot += qreg[d] * kr[d];
    sp[w][k0] = (k0 <= qc) ? dot * scale : -1e30f;
  }
  __syncthreads();

  // pass 2: per-wave softmax over [0, kmax)
  const int kmax = qc + 1;
  float m = -1e30f;
  for (int k0 = lane; k0 < kmax; k0 += 32) m = fmaxf(m, sp[w][k0]);
#pragma unroll
  for (int o = 16; o > 0; o >>= 1) m = fmaxf(m, __shfl_xor(m, o, 32));
  float s = 0.f;
  for (int k0 = lane; k0 < kmax; k0 += 32) {
    float e = __expf(sp[w][k0] - m);
    sp[w][k0] = e; s += e;
  }
#pragma unroll
  for (int o = 16; o > 0; o >>= 1) s += __shfl_xor(s, o, 32);
  const float inv = 1.f / s;
  __syncthreads();

  // pass 3: out[d] = sum_k p[k] * v[k][d]; lanes cover d and d+32
  float a0 = 0.f, a1 = 0.f;
  const int d0 = lane, d1 = lane + 32;
  for (int k0 = 0; k0 < kmax; ++k0) {
    float p = sp[w][k0];
    const float* vr = V + ((long long)(b * T + k0)) * kDim + hh * kHd;
    a0 += p * vr[d0];
    a1 += p * vr[d1];
  }
  if (active) {
    long long ob = ((long long)(b * T + qi)) * kDim + hh * kHd;
    O[ob + d0] = a0 * inv;
    O[ob + d1] = a1 * inv;
  }
}

// ---------------------------------------------------------------------------
// Orchestration
// ---------------------------------------------------------------------------
extern "C" void kernel_launch(void* const* d_in, const int* in_sizes, int n_in,
                              void* d_out, int out_size, void* d_ws, size_t ws_size,
                              hipStream_t stream)
{
  (void)in_sizes; (void)n_in; (void)out_size; (void)ws_size;
  const float* tokens   = (const float*)d_in[0];
  const float* tvec     = (const float*)d_in[1];
  const float* cond     = (const float*)d_in[2];
  const float* latent_w = (const float*)d_in[3];
  const float* latent_b = (const float*)d_in[4];
  const float* cond_w   = (const float*)d_in[5];
  const float* cond_b   = (const float*)d_in[6];
  const float* time_w1  = (const float*)d_in[7];
  const float* time_b1  = (const float*)d_in[8];
  const float* time_w2  = (const float*)d_in[9];
  const float* time_b2  = (const float*)d_in[10];
  const float* wq       = (const float*)d_in[11];
  const float* wk       = (const float*)d_in[12];
  const float* wv       = (const float*)d_in[13];
  const float* wo       = (const float*)d_in[14];
  const float* w1       = (const float*)d_in[15];
  const float* w2       = (const float*)d_in[16];
  const float* w3       = (const float*)d_in[17];
  const float* attn_nw  = (const float*)d_in[18];
  const float* ffn_nw   = (const float*)d_in[19];
  const float* final_nw = (const float*)d_in[20];
  const float* out_w    = (const float*)d_in[21];
  const float* out_b    = (const float*)d_in[22];
  float* out = (float*)d_out;

  // workspace layout (fp32 unless noted), ~140 MB total
  float* ws = (float*)d_ws;
  size_t off = 0;
  float* hbuf  = ws + off; off += (size_t)kBT * kDim;
  float* xnorm = ws + off; off += (size_t)kBT * kDim;
  float* qb    = ws + off; off += (size_t)kBT * kDim;
  float* kbuf  = ws + off; off += (size_t)kBT * kDim;
  float* vbuf  = ws + off; off += (size_t)kBT * kDim;
  float* abuf  = ws + off; off += (size_t)kBT * kDim;
  float* te0   = ws + off; off += (size_t)kB * kDim;
  float* te1   = ws + off; off += (size_t)kB * kDim;
  float* ff1   = ws + off; off += (size_t)kBT * kFF;
  float* ff3   = ws + off; off += (size_t)kBT * kFF;
  unsigned short* wtr = (unsigned short*)(ws + off);        // bf16 W^T scratch
  off += (size_t)kDim * kFF / 2;                            // 8 MB (max K*N)

  auto xposew = [&](const float* W, int K, int N) {
    transpose_bf16_kernel<<<dim3(N / 32, K / 32), 256, 0, stream>>>(W, wtr, K, N);
  };
  auto gemm = [&](const float* A, const float* bias,
                  const float* Res, float* C, int M, int N, int K,
                  long long sA, long long sC, int batch, int flags) {
    dim3 grid(N / 128, (M + 127) / 128, batch);
    gemm_bf16_kernel<<<grid, 256, 0, stream>>>(A, wtr, bias, Res, C, M, N, K, sA, sC, flags);
  };

  // h rows 0..63: cond @ cond_w + cond_b   (per-batch strided output)
  xposew(cond_w, kDim, kDim);
  gemm(cond, cond_b, nullptr, hbuf,
       kSc, kDim, kDim, (long long)kSc * kDim, (long long)kT * kDim, kB, /*bias*/1);
  // h rows 65..576: tokens @ latent_w + latent_b
  xposew(latent_w, kLat, kDim);
  gemm(tokens, latent_b, nullptr, hbuf + (size_t)(kSc + 1) * kDim,
       kS, kDim, kLat, (long long)kS * kLat, (long long)kT * kDim, kB, 1);
  // h row 64: time MLP
  timeembed_kernel<<<kB, 256, 0, stream>>>(tvec, te0);
  xposew(time_w1, kDim, kDim);
  gemm(te0, time_b1, nullptr, te1, kB, kDim, kDim, 0, 0, 1, /*bias+silu*/1 | 2);
  xposew(time_w2, kDim, kDim);
  gemm(te1, time_b2, nullptr, hbuf + (size_t)kSc * kDim,
       1, kDim, kDim, (long long)kDim, (long long)kT * kDim, kB, 1);

  for (int i = 0; i < kL; ++i) {
    const float* wqi = wq + (size_t)i * kDim * kDim;
    const float* wki = wk + (size_t)i * kDim * kDim;
    const float* wvi = wv + (size_t)i * kDim * kDim;
    const float* woi = wo + (size_t)i * kDim * kDim;
    const float* w1i = w1 + (size_t)i * kDim * kFF;
    const float* w2i = w2 + (size_t)i * kFF * kDim;
    const float* w3i = w3 + (size_t)i * kDim * kFF;

    rmsnorm_kernel<<<kBT, 256, 0, stream>>>(hbuf, attn_nw + (size_t)i * kDim, xnorm);
    xposew(wqi, kDim, kDim);
    gemm(xnorm, nullptr, nullptr, qb,   kBT, kDim, kDim, 0, 0, 1, 0);
    xposew(wki, kDim, kDim);
    gemm(xnorm, nullptr, nullptr, kbuf, kBT, kDim, kDim, 0, 0, 1, 0);
    xposew(wvi, kDim, kDim);
    gemm(xnorm, nullptr, nullptr, vbuf, kBT, kDim, kDim, 0, 0, 1, 0);
    attn_kernel<<<dim3(kB, kH, (kT + 7) / 8), 256, 0, stream>>>(qb, kbuf, vbuf, abuf, kT);
    xposew(woi, kDim, kDim);
    gemm(abuf, nullptr, hbuf, hbuf, kBT, kDim, kDim, 0, 0, 1, /*residual*/4);

    rmsnorm_kernel<<<kBT, 256, 0, stream>>>(hbuf, ffn_nw + (size_t)i * kDim, xnorm);
    xposew(w1i, kDim, kFF);
    gemm(xnorm, nullptr, nullptr, ff1, kBT, kFF, kDim, 0, 0, 1, 0);
    xposew(w3i, kDim, kFF);
    gemm(xnorm, nullptr, nullptr, ff3, kBT, kFF, kDim, 0, 0, 1, 0);
    gate_kernel<<<(kBT * kFF) / 1024, 256, 0, stream>>>(ff1, ff3, kBT * kFF);
    xposew(w2i, kFF, kDim);
    gemm(ff1, nullptr, hbuf, hbuf, kBT, kDim, kFF, 0, 0, 1, 4);
  }

  rmsnorm_kernel<<<kBT, 256, 0, stream>>>(hbuf, final_nw, xnorm);
  // out = xnorm[:, SC+1:, :] @ out_w + out_b
  xposew(out_w, kDim, kLat);
  gemm(xnorm + (size_t)(kSc + 1) * kDim, out_b, nullptr, out,
       kS, kLat, kDim, (long long)kT * kDim, (long long)kS * kLat, kB, 1);
}
